// CausalAttention_33724083208915
// MI455X (gfx1250) — matrix operand order repaired and checked by reading, not compiled, not run on hardware
//
#include <hip/hip_runtime.h>
#include <hip/hip_bf16.h>

// ---------------------------------------------------------------------------
// RFA causal attention for MI455X (gfx1250), wave32 + WMMA bf16 (f32 accum)
// + Tensor Data Mover (TDM) staging of scan chunks into LDS.
//   T=2048 B=8 E=1024 H=16 D=64 K=64 (phi dim 2K=128), TAU=1, EPS=1e-6
// Pipeline:
//   1) q/k/v projections: fp32 x @ W^T + b -> bf16    (v_wmma_f32_16x16x32_bf16)
//   2) phi: per-head proj + sin/cos -> bf16 phi_q/phi_k
//   3) chunked causal scan (C=32), fp32 state in LDS, WMMA everywhere,
//      chunk staging via tensor_load_to_lds (TENSORcnt)
//   4) output projection: bf16 attn @ Wo^T + bo -> fp32 out
// ---------------------------------------------------------------------------

typedef __attribute__((ext_vector_type(16))) __bf16    v16bf;
typedef __attribute__((ext_vector_type(8)))  float     v8f;
typedef __attribute__((ext_vector_type(4)))  unsigned  v4u;
typedef __attribute__((ext_vector_type(8)))  unsigned  v8u;

#define WMMA_BF16(A_, B_, C_) \
  __builtin_amdgcn_wmma_f32_16x16x32_bf16(false, (A_), false, (B_), (short)0, (C_), false, false)

constexpr int Tt   = 2048;
constexpr int Bb   = 8;
constexpr int Ee   = 1024;
constexpr int Hh   = 16;
constexpr int Dd   = 64;
constexpr int Kk   = 64;
constexpr int PHID = 2 * Kk;        // 128
constexpr int NROW = Tt * Bb;       // 16384
constexpr float EPSv = 1e-6f;

// ---------------- fragment loaders (k = 16*(lane>>4) + i per lane, wave32) -------------

// A fragment 16x32 (MxK) from row-major memory: p[m*ld + k]
template <typename T>
__device__ __forceinline__ v16bf frag_a_rm(const T* p, int ld) {
  const int l = threadIdx.x & 31;
  const int m = l & 15, kh = l >> 4;
  v16bf f;
#pragma unroll
  for (int i = 0; i < 16; ++i) {
    const int k = (kh << 4) + i;
    f[i] = (__bf16)p[m * ld + k];
  }
  return f;
}

// A fragment 16x32 (MxK) where memory is KxM row-major: A[m][k] = p[k*ld + m]
__device__ __forceinline__ v16bf frag_a_tr(const __bf16* p, int ld) {
  const int l = threadIdx.x & 31;
  const int m = l & 15, kh = l >> 4;
  v16bf f;
#pragma unroll
  for (int i = 0; i < 16; ++i) {
    const int k = (kh << 4) + i;
    f[i] = p[k * ld + m];
  }
  return f;
}

// B fragment 32x16 (KxN) from KxN row-major memory: B[k][n] = p[k*ld + n]
template <typename T>
__device__ __forceinline__ v16bf frag_b_kn(const T* p, int ld) {
  const int l = threadIdx.x & 31;
  const int n = l & 15, kh = l >> 4;
  v16bf f;
#pragma unroll
  for (int i = 0; i < 16; ++i) f[i] = (__bf16)p[((kh << 4) + i) * ld + n];
  return f;
}

// B fragment 32x16 (KxN) from NxK row-major memory: B[k][n] = p[n*ld + k]
// (contiguous in k per lane -> coalesces; used for W (y = x @ W^T) and rm)
template <typename T>
__device__ __forceinline__ v16bf frag_b_nk(const T* p, int ld) {
  const int l = threadIdx.x & 31;
  const int n = l & 15, kh = l >> 4;
  v16bf f;
#pragma unroll
  for (int i = 0; i < 16; ++i) f[i] = (__bf16)p[n * ld + (kh << 4) + i];
  return f;
}

// f32 16x16 C/D accumulator <-> row-major fp32 memory (M = r + 8*(lane>>4))
__device__ __forceinline__ v8f load_acc(const float* p, int ld) {
  const int l = threadIdx.x & 31;
  const int n = l & 15, mh = (l >> 4) << 3;
  v8f c;
#pragma unroll
  for (int r = 0; r < 8; ++r) c[r] = p[(mh + r) * ld + n];
  return c;
}
__device__ __forceinline__ void store_acc(float* p, int ld, v8f c) {
  const int l = threadIdx.x & 31;
  const int n = l & 15, mh = (l >> 4) << 3;
#pragma unroll
  for (int r = 0; r < 8; ++r) p[(mh + r) * ld + n] = c[r];
}

// ---------------- TDM: 2-D tile load, global -> LDS (bf16 elements) ---------
// Builds a D# (groups 0+1, 2-D form: VADDR2/3 = NULL) per CDNA5 ISA and
// issues tensor_load_to_lds. One issue per wave; tracked with TENSORcnt.
//   tile_w elements per row (bf16), tile_h rows, row stride = stride elements.
__device__ __forceinline__ void tdm_load_2d(unsigned lds_addr, const void* gptr,
                                            unsigned tile_w, unsigned tile_h,
                                            unsigned stride_elems) {
  const unsigned long long ga = (unsigned long long)gptr;
  v4u g0;
  g0[0] = 1u;                                           // count=1, user-mode D#
  g0[1] = lds_addr;                                     // lds_addr [63:32]
  g0[2] = (unsigned)(ga & 0xFFFFFFFFu);                 // global_addr[31:0]
  g0[3] = (unsigned)((ga >> 32) & 0x1FFFFFFu)           // global_addr[56:32]
        | (2u << 30);                                   // type=2 ("image")
  v8u g1;
  g1[0] = 1u << 16;                                     // data_size=1 (2 bytes)
  g1[1] = (tile_w & 0xFFFFu) << 16;                     // tensor_dim0[15:0]
  g1[2] = ((tile_w >> 16) & 0xFFFFu)                    // tensor_dim0[31:16]
        | ((tile_h & 0xFFFFu) << 16);                   // tensor_dim1[15:0]
  g1[3] = ((tile_h >> 16) & 0xFFFFu)                    // tensor_dim1[31:16]
        | ((tile_w & 0xFFFFu) << 16);                   // tile_dim0
  g1[4] = tile_h & 0xFFFFu;                             // tile_dim1 (tile_dim2=0)
  g1[5] = stride_elems;                                 // tensor_dim0_stride[31:0]
  g1[6] = 0;                                            // stride[47:32], dim1_stride lo
  g1[7] = 0;
  asm volatile("tensor_load_to_lds %0, %1" :: "s"(g0), "s"(g1) : "memory");
}

// ---------------- kernel 1/4: Y = (X @ W^T + bias) * scale ------------------
// X: [N, Ein] (fp32 or bf16), W: [Eout, Ein] fp32, Y: [N, Eout] (bf16 or fp32)
// one wave -> 16x64 output (A fragment reused across 4 B tiles)

template <typename AT, typename OT>
__global__ __launch_bounds__(256) void wmma_gemm_xWT(
    const AT* __restrict__ X, const float* __restrict__ W,
    const float* __restrict__ bias, OT* __restrict__ Y,
    int N, int Ein, int Eout, float scale) {
  const int wave = threadIdx.x >> 5;
  const int lane = threadIdx.x & 31;
  const int tilesX = Eout >> 6;
  const int tile = blockIdx.x * 8 + wave;
  if (tile >= (N >> 4) * tilesX) return;
  const int row0 = (tile / tilesX) << 4;
  const int col0 = (tile % tilesX) << 6;

  v8f acc0{}, acc1{}, acc2{}, acc3{};
  for (int kk = 0; kk < Ein; kk += 32) {
    v16bf a  = frag_a_rm(X + (size_t)row0 * Ein + kk, Ein);
    v16bf b0 = frag_b_nk(W + (size_t)(col0 +  0) * Ein + kk, Ein);
    v16bf b1 = frag_b_nk(W + (size_t)(col0 + 16) * Ein + kk, Ein);
    v16bf b2 = frag_b_nk(W + (size_t)(col0 + 32) * Ein + kk, Ein);
    v16bf b3 = frag_b_nk(W + (size_t)(col0 + 48) * Ein + kk, Ein);
    acc0 = WMMA_BF16(a, b0, acc0);
    acc1 = WMMA_BF16(a, b1, acc1);
    acc2 = WMMA_BF16(a, b2, acc2);
    acc3 = WMMA_BF16(a, b3, acc3);
  }

  const int n16 = lane & 15;
  const int mh  = (lane >> 4) << 3;
#pragma unroll
  for (int j = 0; j < 4; ++j) {
    v8f acc = (j == 0) ? acc0 : (j == 1) ? acc1 : (j == 2) ? acc2 : acc3;
    const int col = col0 + j * 16 + n16;
    const float bvv = bias[col];
#pragma unroll
    for (int r = 0; r < 8; ++r) {
      const float val = (acc[r] + bvv) * scale;
      Y[(size_t)(row0 + mh + r) * Eout + col] = (OT)val;
    }
  }
}

// ---------------- kernel 2: phi = [sin, cos](q_h @ rm_h^T) * K^-1/2 ---------
// Qh: [N, E] bf16 (head-major inner layout, h*64+d); RM: [H, K, D] fp32
// Phi: [N, H*2K] bf16. One wave -> 16 rows x one full head (K=64 proj cols).

__global__ __launch_bounds__(256) void phi_kernel(
    const __bf16* __restrict__ Qh, const float* __restrict__ RM,
    __bf16* __restrict__ Phi) {
  const int wave = threadIdx.x >> 5;
  const int lane = threadIdx.x & 31;
  const int tile = blockIdx.x * 8 + wave;   // (NROW/16)*H tiles
  const int h  = tile & (Hh - 1);
  const int rt = tile >> 4;
  if (rt >= (NROW >> 4)) return;
  const int row0 = rt << 4;

  v8f p0{}, p1{}, p2{}, p3{};
  for (int kk = 0; kk < Dd; kk += 32) {
    v16bf a  = frag_a_rm(Qh + (size_t)row0 * Ee + h * Dd + kk, Ee);
    v16bf b0 = frag_b_nk(RM + (size_t)h * Kk * Dd +  0 * Dd + kk, Dd);
    v16bf b1 = frag_b_nk(RM + (size_t)h * Kk * Dd + 16 * Dd + kk, Dd);
    v16bf b2 = frag_b_nk(RM + (size_t)h * Kk * Dd + 32 * Dd + kk, Dd);
    v16bf b3 = frag_b_nk(RM + (size_t)h * Kk * Dd + 48 * Dd + kk, Dd);
    p0 = WMMA_BF16(a, b0, p0);
    p1 = WMMA_BF16(a, b1, p1);
    p2 = WMMA_BF16(a, b2, p2);
    p3 = WMMA_BF16(a, b3, p3);
  }

  const int n16 = lane & 15;
  const int mh  = (lane >> 4) << 3;
  const float ksc = 0.125f;   // K^-0.5 (1/TAU == 1 folded in)
#pragma unroll
  for (int j = 0; j < 4; ++j) {
    v8f acc = (j == 0) ? p0 : (j == 1) ? p1 : (j == 2) ? p2 : p3;
    const int kidx = j * 16 + n16;
#pragma unroll
    for (int r = 0; r < 8; ++r) {
      float s, c;
      __sincosf(acc[r], &s, &c);
      const size_t base = (size_t)(row0 + mh + r) * (Hh * PHID) + h * PHID;
      Phi[base + kidx]      = (__bf16)(s * ksc);
      Phi[base + Kk + kidx] = (__bf16)(c * ksc);
    }
  }
}

// ---------------- kernel 3: chunked causal RFA scan -------------------------
// One workgroup per (b,h). fp32 state S[2K][D] + z[2K] lives in LDS; chunk
// C=32, staged by the Tensor Data Mover. Per chunk:
//   A  = causal_mask(Qc @ Kc^T)                (WMMA, bf16-ified into LDS)
//   qz = Qc @ z_prev + rowsum(A), clamped
//   O  = Qc @ S_prev + A @ Vc ; out = O / qz   (WMMA, S fed as bf16 on the fly)
//   S += Kc^T @ Vc  (WMMA with fp32 C/D from LDS -> fp32 accumulation)
//   z += colsum(Kc)

__global__ __launch_bounds__(256) void rfa_scan(
    const __bf16* __restrict__ PhiQ, const __bf16* __restrict__ PhiK,
    const __bf16* __restrict__ Vb, __bf16* __restrict__ Attn) {
  constexpr int C = 32;
  __shared__ __align__(16) float  S[PHID][Dd];    // 32 KB fp32 state
  __shared__ __align__(16) float  zst[PHID];
  __shared__ __align__(16) __bf16 Qc[C][PHID];    // 8 KB
  __shared__ __align__(16) __bf16 Kc[C][PHID];    // 8 KB
  __shared__ __align__(16) __bf16 Vc[C][Dd];      // 4 KB
  __shared__ __align__(16) __bf16 Am[C][C];       // 2 KB
  __shared__ __align__(16) float  qz[C];

  const int tid  = threadIdx.x;
  const int wave = tid >> 5;
  const int lane = tid & 31;
  const int bb   = blockIdx.x / Hh;
  const int h    = blockIdx.x % Hh;
  const int n16  = lane & 15;
  const int mh   = (lane >> 4) << 3;

  // LDS byte offsets of the staging tiles (uniform; aperture keeps offset in
  // the low 32 bits of the generic address)
  const unsigned ldsQ = (unsigned)(unsigned long long)(uintptr_t)&Qc[0][0];
  const unsigned ldsK = (unsigned)(unsigned long long)(uintptr_t)&Kc[0][0];
  const unsigned ldsV = (unsigned)(unsigned long long)(uintptr_t)&Vc[0][0];

  for (int i = tid; i < PHID * Dd; i += 256) (&S[0][0])[i] = 0.f;
  if (tid < PHID) zst[tid] = 0.f;
  __syncthreads();

  for (int c0 = 0; c0 < Tt; c0 += C) {
    // ---- stage chunk via Tensor Data Mover (waves 0..2 issue one D# each);
    //      waves 5..7 prefetch next chunk rows into L2 ----
    {
      const size_t gphi = ((size_t)c0 * Bb + bb) * (Hh * PHID) + (size_t)h * PHID;
      const size_t gv   = ((size_t)c0 * Bb + bb) * Ee + (size_t)h * Dd;
      if (wave == 0) {
        tdm_load_2d(ldsQ, PhiQ + gphi, PHID, C, Bb * Hh * PHID);
      } else if (wave == 1) {
        tdm_load_2d(ldsK, PhiK + gphi, PHID, C, Bb * Hh * PHID);
      } else if (wave == 2) {
        tdm_load_2d(ldsV, Vb + gv, Dd, C, Bb * Ee);
      } else if (wave >= 5 && c0 + C < Tt) {
        const size_t n1 = (size_t)(c0 + C + (tid & 31)) * Bb + bb;
        if (wave == 5)      __builtin_prefetch(PhiQ + n1 * (Hh * PHID) + h * PHID, 0, 0);
        else if (wave == 6) __builtin_prefetch(PhiK + n1 * (Hh * PHID) + h * PHID, 0, 0);
        else                __builtin_prefetch(Vb + n1 * Ee + h * Dd, 0, 0);
      }
      __builtin_amdgcn_s_wait_tensorcnt(0);
    }
    __syncthreads();

    // ---- A = mask(Qc @ Kc^T) on waves 0..3; qz inter-part on wave 4 ----
    if (wave < 4) {
      const int ti = wave >> 1, tj = wave & 1;
      v8f acc{};
      for (int kk = 0; kk < PHID; kk += 32) {
        v16bf a = frag_a_rm(&Qc[ti * 16][kk], PHID);
        v16bf b = frag_b_nk(&Kc[tj * 16][kk], PHID);   // B[k][n] = Kc[n][k]
        acc = WMMA_BF16(a, b, acc);
      }
#pragma unroll
      for (int r = 0; r < 8; ++r) {
        const int i = ti * 16 + mh + r, j = tj * 16 + n16;
        Am[i][j] = (__bf16)((j <= i) ? acc[r] : 0.f);
      }
    } else {
      const int i = tid - 128;
      if (i < C) {
        float a = 0.f;
        for (int k = 0; k < PHID; ++k) a += (float)Qc[i][k] * zst[k];
        qz[i] = a;
      }
    }
    __syncthreads();

    // ---- qz += rowsum(masked A); clamp ----
    if (tid < C) {
      float a = qz[tid];
      for (int j = 0; j <= tid; ++j) a += (float)Am[tid][j];
      qz[tid] = fmaxf(a, EPSv);
    }
    __syncthreads();

    // ---- O = Qc @ S_prev + Am @ Vc; one 16x16 tile per wave (2x4) ----
    {
      const int ti = wave >> 2, tj = wave & 3;
      v8f acc{};
      for (int kk = 0; kk < PHID; kk += 32) {
        v16bf a = frag_a_rm(&Qc[ti * 16][kk], PHID);
        v16bf b = frag_b_kn(&S[kk][tj * 16], Dd);      // fp32 -> bf16 on the fly
        acc = WMMA_BF16(a, b, acc);
      }
      {
        v16bf a = frag_a_rm(&Am[ti * 16][0], C);
        v16bf b = frag_b_kn(&Vc[0][tj * 16], Dd);
        acc = WMMA_BF16(a, b, acc);
      }
#pragma unroll
      for (int r = 0; r < 8; ++r) {
        const int i = ti * 16 + mh + r, d = tj * 16 + n16;
        const float val = acc[r] / qz[i];
        Attn[((size_t)(c0 + i) * Bb + bb) * Ee + h * Dd + d] = (__bf16)val;
      }
    }
    __syncthreads();

    // ---- S += Kc^T @ Vc (8x4 tiles, 4/wave, fp32 C/D); z += colsum(Kc) ----
#pragma unroll
    for (int t = 0; t < 4; ++t) {
      const int tile = wave * 4 + t;
      const int tm = tile >> 2, tn = tile & 3;
      v8f acc = load_acc(&S[tm * 16][tn * 16], Dd);
      v16bf a = frag_a_tr(&Kc[0][tm * 16], PHID);      // A[m][k] = Kc[k][m]
      v16bf b = frag_b_kn(&Vc[0][tn * 16], Dd);
      acc = WMMA_BF16(a, b, acc);
      store_acc(&S[tm * 16][tn * 16], Dd, acc);
    }
    if (tid < PHID) {
      float sum = 0.f;
      for (int j = 0; j < C; ++j) sum += (float)Kc[j][tid];
      zst[tid] += sum;
    }
    __syncthreads();
  }
}

// ---------------------------------------------------------------------------

extern "C" void kernel_launch(void* const* d_in, const int* in_sizes, int n_in,
                              void* d_out, int out_size, void* d_ws, size_t ws_size,
                              hipStream_t stream) {
  const float* x  = (const float*)d_in[0];
  const float* rm = (const float*)d_in[1];
  const float* Wq = (const float*)d_in[2];
  const float* bq = (const float*)d_in[3];
  const float* Wk = (const float*)d_in[4];
  const float* bk = (const float*)d_in[5];
  const float* Wv = (const float*)d_in[6];
  const float* bv = (const float*)d_in[7];
  const float* Wo = (const float*)d_in[8];
  const float* bo = (const float*)d_in[9];
  float* out = (float*)d_out;

  char* ws = (char*)d_ws;
  size_t off = 0;
  __bf16* qb   = (__bf16*)(ws + off); off += (size_t)NROW * Ee * 2;          // 32MB
  __bf16* kb   = (__bf16*)(ws + off); off += (size_t)NROW * Ee * 2;          // 32MB
  __bf16* vb   = (__bf16*)(ws + off); off += (size_t)NROW * Ee * 2;          // 32MB
  __bf16* phiq = (__bf16*)(ws + off); off += (size_t)NROW * Hh * PHID * 2;   // 64MB
  __bf16* phik = (__bf16*)(ws + off); off += (size_t)NROW * Hh * PHID * 2;   // 64MB
  __bf16* attn = (__bf16*)(ws + off); off += (size_t)NROW * Ee * 2;          // 32MB
  (void)ws_size; (void)in_sizes; (void)n_in; (void)out_size;

  const dim3 blk(256);
  const int gemmBlocks = (NROW / 16) * (Ee / 64) / 8;   // 2048
  const int phiBlocks  = (NROW / 16) * Hh / 8;          // 2048
  const float qscale = 0.125f;                          // D^-0.5

  wmma_gemm_xWT<float, __bf16><<<gemmBlocks, blk, 0, stream>>>(x, Wq, bq, qb, NROW, Ee, Ee, qscale);
  wmma_gemm_xWT<float, __bf16><<<gemmBlocks, blk, 0, stream>>>(x, Wk, bk, kb, NROW, Ee, Ee, 1.0f);
  wmma_gemm_xWT<float, __bf16><<<gemmBlocks, blk, 0, stream>>>(x, Wv, bv, vb, NROW, Ee, Ee, 1.0f);
  phi_kernel<<<phiBlocks, blk, 0, stream>>>(qb, rm, phiq);
  phi_kernel<<<phiBlocks, blk, 0, stream>>>(kb, rm, phik);
  rfa_scan<<<Bb * Hh, blk, 0, stream>>>(phiq, phik, vb, attn);
  wmma_gemm_xWT<__bf16, float><<<gemmBlocks, blk, 0, stream>>>(attn, Wo, bo, out, NROW, Ee, Ee, 1.0f);
}
